// KnotNet_12927851561368
// MI455X (gfx1250) — matrix operand order repaired and checked by reading, not compiled, not run on hardware
//
#include <hip/hip_runtime.h>
#include <hip/hip_bf16.h>

// ---------------------------------------------------------------------------
// KnotNet on MI455X (gfx1250, wave32):
//   scan_kernel : 1 wave per batch; 4x64 state in VGPRs (2 cols/lane),
//                 braids preloaded to VGPRs and broadcast with __shfl,
//                 cos/sin lookup tables (7 entries) in LDS, wave-shuffle LN.
//   gemm1/gemm2 : f16 WMMA (v_wmma_f32_16x16x32_f16), f32 accumulate.
//                 Weights staged in LDS via async global->LDS (ASYNCcnt),
//                 A fragments held fully in VGPRs, B fragments from ds_load.
//   head_kernel : scalar 64x2 head + sigmoid.
// ---------------------------------------------------------------------------

typedef __attribute__((ext_vector_type(16))) _Float16 v16h;
typedef __attribute__((ext_vector_type(8)))  float    v8f;

union Frag16 { v16h h; uint4 q[2]; };

#define BATCH   4096
#define TSTEPS  256
#define LN_EPS  1e-5f

__device__ __forceinline__ float sigmoidf_dev(float x) {
    return 1.0f / (1.0f + __expf(-x));
}

__device__ __forceinline__ float wave_sum(float x) {
    #pragma unroll
    for (int off = 16; off > 0; off >>= 1) x += __shfl_xor(x, off, 32);
    return x;
}

// gfx1250 async global->LDS copy, 16 bytes per lane. Tracked by ASYNCcnt.
__device__ __forceinline__ void async_g2l_b128(uint32_t lds_byte_addr,
                                               const void* gptr) {
    asm volatile("global_load_async_to_lds_b128 %0, %1, off"
                 :: "v"(lds_byte_addr), "v"((uint64_t)(uintptr_t)gptr)
                 : "memory");
}
__device__ __forceinline__ void wait_async_zero() {
    asm volatile("s_wait_asynccnt 0x0" ::: "memory");
}

// ---------------------------------------------------------------------------
// Weight prep: transpose w1 (256x128) -> w1t (128x256 f16, N-major),
//              transpose w2 (128x64)  -> w2t (64x128 f16, N-major)
// ---------------------------------------------------------------------------
__global__ __launch_bounds__(256) void prep_weights_kernel(
    const float* __restrict__ w1, const float* __restrict__ w2,
    _Float16* __restrict__ w1t, _Float16* __restrict__ w2t)
{
    int i = blockIdx.x * 256 + threadIdx.x;
    if (i < 128 * 256) {            // w1t[n][k] = w1[k][n]
        int n = i >> 8, k = i & 255;
        w1t[i] = (_Float16)w1[k * 128 + n];
    }
    if (i < 64 * 128) {             // w2t[n][k] = w2[k][n]
        int n = i >> 7, k = i & 127;
        w2t[i] = (_Float16)w2[k * 64 + n];
    }
}

// ---------------------------------------------------------------------------
// Scan: 8 waves/block, 1 batch per wave. Lane L owns columns L and L+32.
// ---------------------------------------------------------------------------
__global__ __launch_bounds__(256) void scan_kernel(
    const int*   __restrict__ braids,        // (B, T)
    const float* __restrict__ initial_state, // (4, 64)
    const float* __restrict__ st1,           // (1,)
    const float* __restrict__ st2,           // (2,)
    const float* __restrict__ meta,          // (3,)
    const float* __restrict__ gates,         // (3,)
    const float* __restrict__ lnw,           // (64,)
    const float* __restrict__ lnb,           // (64,)
    _Float16*    __restrict__ Aout)          // (B, 256) f16
{
    __shared__ float cT[8], sT[8];

    const int lane = threadIdx.x & 31;
    const int wave = threadIdx.x >> 5;
    const int b    = blockIdx.x * 8 + wave;

    // Build cos/sin tables for braid values g in [-3, 3]
    if (threadIdx.x < 7) {
        int g = (int)threadIdx.x - 3;
        float th = 0.0f;
        if (g != 0) {
            int p = (g < 0 ? -g : g) - 1;                 // 0..2
            float theta = (p == 0) ? st1[0] : ((p == 1) ? st2[0] : st2[1]);
            float gv = sigmoidf_dev(gates[p]);
            th = (g < 0 ? -1.0f : 1.0f) * theta * gv;
        }
        cT[threadIdx.x] = __cosf(th);
        sT[threadIdx.x] = __sinf(th);
    }
    __syncthreads();

    // State: 4 rows x 2 columns per lane
    const int c0 = lane, c1 = lane + 32;
    float s00 = initial_state[0 * 64 + c0], s01 = initial_state[0 * 64 + c1];
    float s10 = initial_state[1 * 64 + c0], s11 = initial_state[1 * 64 + c1];
    float s20 = initial_state[2 * 64 + c0], s21 = initial_state[2 * 64 + c1];
    float s30 = initial_state[3 * 64 + c0], s31 = initial_state[3 * 64 + c1];

    // Preload this batch's braids: 8 ints per lane
    int mb0 = braids[b * TSTEPS +   0 + lane];
    int mb1 = braids[b * TSTEPS +  32 + lane];
    int mb2 = braids[b * TSTEPS +  64 + lane];
    int mb3 = braids[b * TSTEPS +  96 + lane];
    int mb4 = braids[b * TSTEPS + 128 + lane];
    int mb5 = braids[b * TSTEPS + 160 + lane];
    int mb6 = braids[b * TSTEPS + 192 + lane];
    int mb7 = braids[b * TSTEPS + 224 + lane];

    #define ROT(u0, v0, u1, v1, c, sn)                       \
        { float nu0 = u0 * c - v0 * sn, nv0 = u0 * sn + v0 * c; \
          float nu1 = u1 * c - v1 * sn, nv1 = u1 * sn + v1 * c; \
          u0 = nu0; v0 = nv0; u1 = nu1; v1 = nv1; }

    #define STEP_CHUNK(reg)                                    \
        for (int i = 0; i < 32; ++i) {                         \
            int g = __shfl(reg, i, 32);                        \
            if (g != 0) {                                      \
                float c = cT[g + 3], sn = sT[g + 3];           \
                int p = (g < 0 ? -g : g) - 1;                  \
                if (p == 0)      { ROT(s00, s10, s01, s11, c, sn) } \
                else if (p == 1) { ROT(s10, s20, s11, s21, c, sn) } \
                else             { ROT(s20, s30, s21, s31, c, sn) } \
            }                                                  \
        }

    STEP_CHUNK(mb0) STEP_CHUNK(mb1) STEP_CHUNK(mb2) STEP_CHUNK(mb3)
    STEP_CHUNK(mb4) STEP_CHUNK(mb5) STEP_CHUNK(mb6) STEP_CHUNK(mb7)

    // Meta rotations: planes (0,1),(1,2),(2,3)
    {
        float gv0 = sigmoidf_dev(gates[0]);
        float gv1 = sigmoidf_dev(gates[1]);
        float gv2 = sigmoidf_dev(gates[2]);
        float th0 = meta[0] * gv0, th1 = meta[1] * gv1, th2 = meta[2] * gv2;
        float c, sn;
        c = __cosf(th0); sn = __sinf(th0); ROT(s00, s10, s01, s11, c, sn)
        c = __cosf(th1); sn = __sinf(th1); ROT(s10, s20, s11, s21, c, sn)
        c = __cosf(th2); sn = __sinf(th2); ROT(s20, s30, s21, s31, c, sn)
    }

    // LayerNorm per row over H=64, then write f16 activations
    const float w0 = lnw[c0], w1v = lnw[c1];
    const float bb0 = lnb[c0], bb1 = lnb[c1];
    _Float16* dst = Aout + b * 256;

    #define LN_ROW(r, x0, x1)                                          \
        { float mu = wave_sum(x0 + x1) * (1.0f / 64.0f);               \
          float d0 = x0 - mu, d1 = x1 - mu;                            \
          float var = wave_sum(d0 * d0 + d1 * d1) * (1.0f / 64.0f);    \
          float inv = rsqrtf(var + LN_EPS);                            \
          dst[r * 64 + c0] = (_Float16)(d0 * inv * w0 + bb0);          \
          dst[r * 64 + c1] = (_Float16)(d1 * inv * w1v + bb1); }

    LN_ROW(0, s00, s01)
    LN_ROW(1, s10, s11)
    LN_ROW(2, s20, s21)
    LN_ROW(3, s30, s31)

    #undef LN_ROW
    #undef STEP_CHUNK
    #undef ROT
}

// ---------------------------------------------------------------------------
// GEMM1: H1 = relu(A(4096x256) @ w1(256x128) + b1), f16 WMMA, f32 accum.
// 4 waves/block, each wave: 16 rows x 128 cols (8 accumulators).
// Full w1t (64 KB) staged in LDS via async global->LDS; A held in VGPRs.
// ---------------------------------------------------------------------------
__global__ __launch_bounds__(128) void gemm1_kernel(
    const _Float16* __restrict__ A,    // 4096 x 256
    const _Float16* __restrict__ Bt,   // 128 x 256 (N-major)
    const float*    __restrict__ bias, // 128
    _Float16*       __restrict__ H1)   // 4096 x 128
{
    __shared__ _Float16 sB[128 * 256];               // 64 KB

    const int lane = threadIdx.x & 31;
    const int wave = threadIdx.x >> 5;
    const int m0   = blockIdx.x * 64 + wave * 16;
    const int lm   = lane & 15;
    const int hi   = lane >> 4;

    // Stage Bt into LDS: 65536 bytes, 128 threads x 16 B => 32 iterations.
    {
        const uint32_t lbase = (uint32_t)(uintptr_t)sB;
        const char* gbase = (const char*)Bt;
        #pragma unroll
        for (int it = 0; it < 32; ++it) {
            int off = (it * 128 + (int)threadIdx.x) * 16;
            async_g2l_b128(lbase + off, gbase + off);
        }
    }

    // While the DMA runs, load all A fragments for this wave (8 x v16h).
    Frag16 a[8];
    {
        const _Float16* ap = A + (m0 + lm) * 256 + hi * 8;
        #pragma unroll
        for (int kt = 0; kt < 8; ++kt) {
            a[kt].q[0] = *(const uint4*)(ap + kt * 32);        // K: +0..7
            a[kt].q[1] = *(const uint4*)(ap + kt * 32 + 16);   // K: +16..23
        }
    }

    wait_async_zero();
    __syncthreads();

    v8f acc[8] = {};
    #pragma unroll
    for (int kt = 0; kt < 8; ++kt) {
        const int k0 = kt * 32;
        #pragma unroll
        for (int nt = 0; nt < 8; ++nt) {
            Frag16 bf;
            const _Float16* bp = sB + (nt * 16 + lm) * 256 + k0 + hi * 8;
            bf.q[0] = *(const uint4*)(bp);       // K = k0 + hi*8 + 0..7
            bf.q[1] = *(const uint4*)(bp + 16);  // K = k0 + 16 + hi*8 + 0..7
            acc[nt] = __builtin_amdgcn_wmma_f32_16x16x32_f16(
                false, a[kt].h, false, bf.h, (short)0, acc[nt], false, false);
        }
    }

    #pragma unroll
    for (int nt = 0; nt < 8; ++nt) {
        #pragma unroll
        for (int r = 0; r < 8; ++r) {
            int row = m0 + r + hi * 8;
            int col = nt * 16 + lm;
            float v = acc[nt][r] + bias[col];
            v = v > 0.0f ? v : 0.0f;
            H1[row * 128 + col] = (_Float16)v;
        }
    }
}

// ---------------------------------------------------------------------------
// GEMM2: H2 = relu(H1(4096x128) @ w2(128x64) + b2), f32 output for head.
// w2t (16 KB) staged in LDS via async global->LDS.
// ---------------------------------------------------------------------------
__global__ __launch_bounds__(128) void gemm2_kernel(
    const _Float16* __restrict__ A,    // 4096 x 128
    const _Float16* __restrict__ Bt,   // 64 x 128 (N-major)
    const float*    __restrict__ bias, // 64
    float*          __restrict__ H2)   // 4096 x 64
{
    __shared__ _Float16 sB[64 * 128];                // 16 KB

    const int lane = threadIdx.x & 31;
    const int wave = threadIdx.x >> 5;
    const int m0   = blockIdx.x * 64 + wave * 16;
    const int lm   = lane & 15;
    const int hi   = lane >> 4;

    // Stage Bt into LDS: 16384 bytes, 128 threads x 16 B => 8 iterations.
    {
        const uint32_t lbase = (uint32_t)(uintptr_t)sB;
        const char* gbase = (const char*)Bt;
        #pragma unroll
        for (int it = 0; it < 8; ++it) {
            int off = (it * 128 + (int)threadIdx.x) * 16;
            async_g2l_b128(lbase + off, gbase + off);
        }
    }

    Frag16 a[4];
    {
        const _Float16* ap = A + (m0 + lm) * 128 + hi * 8;
        #pragma unroll
        for (int kt = 0; kt < 4; ++kt) {
            a[kt].q[0] = *(const uint4*)(ap + kt * 32);
            a[kt].q[1] = *(const uint4*)(ap + kt * 32 + 16);
        }
    }

    wait_async_zero();
    __syncthreads();

    v8f acc[4] = {};
    #pragma unroll
    for (int kt = 0; kt < 4; ++kt) {
        const int k0 = kt * 32;
        #pragma unroll
        for (int nt = 0; nt < 4; ++nt) {
            Frag16 bf;
            const _Float16* bp = sB + (nt * 16 + lm) * 128 + k0 + hi * 8;
            bf.q[0] = *(const uint4*)(bp);
            bf.q[1] = *(const uint4*)(bp + 16);
            acc[nt] = __builtin_amdgcn_wmma_f32_16x16x32_f16(
                false, a[kt].h, false, bf.h, (short)0, acc[nt], false, false);
        }
    }

    #pragma unroll
    for (int nt = 0; nt < 4; ++nt) {
        #pragma unroll
        for (int r = 0; r < 8; ++r) {
            int row = m0 + r + hi * 8;
            int col = nt * 16 + lm;
            float v = acc[nt][r] + bias[col];
            v = v > 0.0f ? v : 0.0f;
            H2[row * 64 + col] = v;
        }
    }
}

// ---------------------------------------------------------------------------
// Head: out0 = sigmoid(h2 @ w3[:,0] + b3[0]); out1 = h2 @ w3[:,1] + b3[1]
// ---------------------------------------------------------------------------
__global__ __launch_bounds__(256) void head_kernel(
    const float* __restrict__ H2,  // 4096 x 64
    const float* __restrict__ w3,  // 64 x 2
    const float* __restrict__ b3,  // 2
    float*       __restrict__ out) // [sigmoid(o0) x 4096, o1 x 4096]
{
    int row = blockIdx.x * 256 + threadIdx.x;
    if (row >= BATCH) return;
    const float* h = H2 + row * 64;
    float o0 = b3[0], o1 = b3[1];
    #pragma unroll
    for (int i = 0; i < 64; ++i) {
        float v = h[i];
        o0 += v * w3[2 * i + 0];
        o1 += v * w3[2 * i + 1];
    }
    out[row]         = sigmoidf_dev(o0);
    out[BATCH + row] = o1;
}

// ---------------------------------------------------------------------------
extern "C" void kernel_launch(void* const* d_in, const int* in_sizes, int n_in,
                              void* d_out, int out_size, void* d_ws, size_t ws_size,
                              hipStream_t stream) {
    const int*   braids = (const int*)  d_in[0];
    const float* init   = (const float*)d_in[1];
    const float* st1    = (const float*)d_in[2];
    const float* st2    = (const float*)d_in[3];
    const float* meta   = (const float*)d_in[4];
    const float* gates  = (const float*)d_in[5];
    const float* lnw    = (const float*)d_in[6];
    const float* lnb    = (const float*)d_in[7];
    const float* w1     = (const float*)d_in[8];
    const float* b1     = (const float*)d_in[9];
    const float* w2     = (const float*)d_in[10];
    const float* b2     = (const float*)d_in[11];
    const float* w3     = (const float*)d_in[12];
    const float* b3     = (const float*)d_in[13];
    float* out = (float*)d_out;

    char* ws = (char*)d_ws;
    _Float16* Aact = (_Float16*)(ws + 0);          // 4096*256*2 = 2 MB
    _Float16* H1   = (_Float16*)(ws + 2097152);    // 4096*128*2 = 1 MB
    float*    H2   = (float*)   (ws + 3145728);    // 4096*64*4  = 1 MB
    _Float16* w1t  = (_Float16*)(ws + 4194304);    // 128*256*2  = 64 KB
    _Float16* w2t  = (_Float16*)(ws + 4259840);    // 64*128*2   = 16 KB

    prep_weights_kernel<<<128, 256, 0, stream>>>(w1, w2, w1t, w2t);
    scan_kernel<<<BATCH / 8, 256, 0, stream>>>(braids, init, st1, st2, meta,
                                               gates, lnw, lnb, Aact);
    gemm1_kernel<<<BATCH / 64, 128, 0, stream>>>(Aact, w1t, b1, H1);
    gemm2_kernel<<<BATCH / 64, 128, 0, stream>>>(H1, w2t, b2, H2);
    head_kernel<<<BATCH / 256, 256, 0, stream>>>(H2, w3, b3, out);
}
